// CrissCrossAttention_23295902614155
// MI455X (gfx1250) — matrix operand order repaired and checked by reading, not compiled, not run on hardware
//
#include <hip/hip_runtime.h>
#include <stdint.h>

#define BB 16
#define CC 256
#define HH 96
#define WW 96
#define KT 11
#define C8 32
#define RLEN (KT*CC)   /* 2816 reduction length for convs */
#define LCOL 192       /* H + W logits */

typedef __attribute__((ext_vector_type(16))) __bf16 v16bf;
typedef __attribute__((ext_vector_type(8)))  float  v8f;

union FragB { uint4 q[2]; v16bf v; };

__device__ __forceinline__ unsigned short f2bf(float x) {
  union { float f; unsigned u; } c; c.f = x;
  unsigned r = c.u + 0x7fffu + ((c.u >> 16) & 1u);   // round-to-nearest-even
  return (unsigned short)(r >> 16);
}

// ---------------------------------------------------------------------------
// Pack x (f32 NCHW) -> bf16 [b][h][w][c] via 32x32 LDS tile transpose:
// coalesced 128B reads along W, coalesced 64B bf16 writes along C.
// ---------------------------------------------------------------------------
__global__ __launch_bounds__(256) void pack_x_kernel(
    const float* __restrict__ x, unsigned short* __restrict__ xt) {
  __shared__ float tile[32][33];               // pad to kill bank conflicts
  unsigned t = blockIdx.x;                     // (b, h, wt, ct), ct innermost
  const unsigned ct = t % 8;  t /= 8;
  const unsigned wt = t % 3;  t /= 3;
  const unsigned h  = t % HH; const unsigned b = t / HH;
  const unsigned c0 = ct * 32, w0 = wt * 32;
  const unsigned lw = threadIdx.x & 31;        // W lane (read)
  const unsigned lr = threadIdx.x >> 5;        // 8 rows per round
  #pragma unroll
  for (int r = 0; r < 4; ++r) {
    const unsigned cc = lr + r * 8;
    tile[cc][lw] = x[((b * CC + c0 + cc) * HH + h) * WW + w0 + lw];
  }
  __syncthreads();
  const unsigned lc = threadIdx.x & 31;        // C lane (write)
  #pragma unroll
  for (int r = 0; r < 4; ++r) {
    const unsigned ww = lr + r * 8;
    xt[((b * HH + h) * WW + w0 + ww) * CC + c0 + lc] = f2bf(tile[lc][ww]);
  }
}

// ---------------------------------------------------------------------------
// Pack conv weights (f32 [co][ci][11] flat) -> bf16 [co][tap*256+ci]
// ---------------------------------------------------------------------------
__global__ __launch_bounds__(256) void pack_w_kernel(
    const float* __restrict__ w, unsigned short* __restrict__ wp, int cout) {
  int i = blockIdx.x * 256 + threadIdx.x;
  if (i >= cout * RLEN) return;
  int r = i % RLEN, co = i / RLEN;
  int tap = r / CC, ci = r % CC;
  wp[i] = f2bf(w[(co * CC + ci) * KT + tap]);
}

// ---------------------------------------------------------------------------
// Implicit-GEMM 1-D conv via v_wmma_f32_16x16x32_bf16, N-register-blocked:
// one wave = 16 pixels x (NB*16) chout; A fragment loaded once per K-chunk,
// reused for NB WMMAs. All divisors compile-time; 32-bit output addressing.
// ---------------------------------------------------------------------------
template<int COUT, int VERT, int NB>
__global__ __launch_bounds__(256) void conv_wmma_kernel(
    const unsigned short* __restrict__ xt, const unsigned short* __restrict__ wp,
    const float* __restrict__ bias, unsigned short* __restrict__ out,
    unsigned sB, unsigned sH, unsigned sW, unsigned sC) {
  constexpr int NG = COUT / (16 * NB);         // chout groups per pixel tile
  const int lane = threadIdx.x & 31;
  const int wave = threadIdx.x >> 5;
  unsigned tile = blockIdx.x * 8u + wave;
  const int cog = tile % NG; tile /= NG;
  const int wt  = tile % (WW / 16); tile /= (WW / 16);
  const int h   = tile % HH;
  const int b   = tile / HH;
  const int w0  = wt * 16;
  const int m   = lane & 15;                   // A row (pixel) / B col (chout)
  const int kb  = (lane & 16) ? 8 : 0;         // A K-half base (ISA 16-bit A layout)
  const int klo = (lane & 16) ? 8 : 0;         // B K-half base (same K order as A)
  const int co0 = cog * 16 * NB;

  const unsigned short* wrow[NB];
  #pragma unroll
  for (int nb = 0; nb < NB; ++nb)
    wrow[nb] = wp + (size_t)(co0 + nb * 16 + m) * RLEN + klo;

  v8f acc[NB];
  #pragma unroll
  for (int nb = 0; nb < NB; ++nb) acc[nb] = (v8f){};

  for (int tap = 0; tap < KT; ++tap) {
    const int hs = VERT ? (h + tap - 5) : h;
    if (VERT && (hs < 0 || hs >= HH)) continue;        // wave-uniform zero-pad skip
    const int wsrc  = VERT ? (w0 + m) : (w0 + m + tap - 5);
    const bool valid = (wsrc >= 0) && (wsrc < WW);
    const int  wc    = valid ? wsrc : 0;
    const unsigned short* pix = xt + (unsigned)((b * HH + hs) * WW + wc) * CC;
    for (int ci0 = 0; ci0 < CC; ci0 += 32) {
      FragB A;
      if (valid) {
        A.q[0] = *(const uint4*)(pix + ci0 + kb);
        A.q[1] = *(const uint4*)(pix + ci0 + kb + 16);
      } else {
        A.q[0] = make_uint4(0u, 0u, 0u, 0u);
        A.q[1] = make_uint4(0u, 0u, 0u, 0u);
      }
      #pragma unroll
      for (int nb = 0; nb < NB; ++nb) {
        FragB Bf;
        const unsigned short* wq = wrow[nb] + tap * CC + ci0;
        Bf.q[0] = *(const uint4*)(wq);
        Bf.q[1] = *(const uint4*)(wq + 16);
        acc[nb] = __builtin_amdgcn_wmma_f32_16x16x32_bf16(
                      false, A.v, false, Bf.v, (short)0, acc[nb], false, false);
      }
    }
  }

  const unsigned mi0 = (lane >> 4) << 3;       // D layout: M = v + 8*(lane>>4)
  #pragma unroll
  for (int nb = 0; nb < NB; ++nb) {
    const unsigned co = co0 + nb * 16 + m;
    const float bv = bias[co];
    const unsigned idx0 = (unsigned)b * sB + (unsigned)h * sH
                        + ((unsigned)w0 + mi0) * sW + co * sC;
    #pragma unroll
    for (int v = 0; v < 8; ++v)
      out[idx0 + (unsigned)v * sW] = f2bf(acc[nb][v] + bv);
  }
}

// ---------------------------------------------------------------------------
// Logits: per (b, fixed col/row) E(96x96) = Q(96x32) * K^T(32x96), one K-chunk.
// Q/K panels (6 KB each) staged to LDS via CDNA5 async-to-LDS DMA.
// Templated so e-index strides/mask fold to literals; 32-bit addressing.
// ---------------------------------------------------------------------------
template<int EM, int EF, int COLOFF, int MASK>
__global__ __launch_bounds__(256) void attn_e_kernel(
    const unsigned short* __restrict__ qT, const unsigned short* __restrict__ kT,
    float* __restrict__ e) {
  __shared__ unsigned short sQ[96 * 32];
  __shared__ unsigned short sK[96 * 32];
  const int tid = threadIdx.x;
  const int p = blockIdx.x;                   // b*96 + fixed coordinate
  const int b = p / 96, f = p % 96;

  const unsigned long long qsrc = (unsigned long long)(const void*)(qT + (size_t)p * 96 * 32);
  const unsigned long long ksrc = (unsigned long long)(const void*)(kT + (size_t)p * 96 * 32);
  // 768 x B128 async transfers; 384 boundary is a multiple of 32 -> saddr wave-uniform
  #pragma unroll
  for (int j = 0; j < 3; ++j) {
    const int idx = j * 256 + tid;
    unsigned long long src; unsigned dst, voff;
    if (idx < 384) {
      src = qsrc; voff = (unsigned)idx * 16u;
      dst = (unsigned)(size_t)(&sQ[0]) + (unsigned)idx * 16u;
    } else {
      src = ksrc; voff = (unsigned)(idx - 384) * 16u;
      dst = (unsigned)(size_t)(&sK[0]) + (unsigned)(idx - 384) * 16u;
    }
    asm volatile("global_load_async_to_lds_b128 %0, %1, %2"
                 :: "v"(dst), "v"(voff), "s"(src) : "memory");
  }
  asm volatile("s_wait_asynccnt 0" ::: "memory");
  __syncthreads();

  const int lane = tid & 31, wave = tid >> 5;
  const int m   = lane & 15;
  const int kb  = (lane & 16) ? 8 : 0;
  const int klo = (lane & 16) ? 8 : 0;

  for (int t = wave; t < 36; t += 8) {        // 6x6 tiles of 16x16
    const int mt = t / 6, nt = t % 6;
    FragB A, Bf;
    const unsigned short* qa = sQ + (mt * 16 + m) * 32;
    A.q[0] = *(const uint4*)(qa + kb);
    A.q[1] = *(const uint4*)(qa + kb + 16);
    const unsigned short* kp = sK + (nt * 16 + m) * 32 + klo;  // lane m == lane n
    Bf.q[0] = *(const uint4*)(kp);
    Bf.q[1] = *(const uint4*)(kp + 16);
    v8f acc = {};
    acc = __builtin_amdgcn_wmma_f32_16x16x32_bf16(
              false, A.v, false, Bf.v, (short)0, acc, false, false);
    #pragma unroll
    for (int v = 0; v < 8; ++v) {
      const unsigned mi = mt * 16 + v + ((lane >> 4) << 3);
      const unsigned ni = nt * 16 + (lane & 15);
      float val = acc[v];
      if (MASK && mi == ni) val = -3.0e38f;               // diag -> -inf
      e[((unsigned)b * 9216u + mi * (unsigned)EM + (unsigned)f * (unsigned)EF) * LCOL
        + (unsigned)COLOFF + ni] = val;
    }
  }
}

// ---------------------------------------------------------------------------
// Joint softmax over 192 logits, wave32 per row, shuffle reductions.
// ---------------------------------------------------------------------------
__global__ __launch_bounds__(256) void softmax_kernel(
    const float* __restrict__ e, unsigned short* __restrict__ ah,
    unsigned short* __restrict__ aw) {
  const int lane = threadIdx.x & 31;
  const unsigned row = blockIdx.x * 8u + (threadIdx.x >> 5);
  const float* er = e + row * LCOL;
  float vals[6]; float mx = -3.4e38f;
  #pragma unroll
  for (int k = 0; k < 6; ++k) { vals[k] = er[lane + 32 * k]; mx = fmaxf(mx, vals[k]); }
  #pragma unroll
  for (int o = 16; o > 0; o >>= 1) mx = fmaxf(mx, __shfl_xor(mx, o, 32));
  float s = 0.f;
  #pragma unroll
  for (int k = 0; k < 6; ++k) { vals[k] = __expf(vals[k] - mx); s += vals[k]; }
  #pragma unroll
  for (int o = 16; o > 0; o >>= 1) s += __shfl_xor(s, o, 32);
  const float inv = __frcp_rn(s);
  #pragma unroll
  for (int k = 0; k < 6; ++k) {
    const int j = lane + 32 * k;
    const unsigned short pb = f2bf(vals[k] * inv);
    if (j < 96) ah[row * 96u + j] = pb;
    else        aw[row * 96u + (j - 96)] = pb;
  }
}

// ---------------------------------------------------------------------------
// out_h: per (b,w)  O1(256x96) = Vh(256x96) x Ah^T(96x96)  -> f32 scratch NCHW
// Wave fixes a c-tile, keeps 6 accumulators (n-tiles), reuses each A frag 6x.
// ---------------------------------------------------------------------------
__global__ __launch_bounds__(256) void out_h_kernel(
    const unsigned short* __restrict__ vh, const unsigned short* __restrict__ ah,
    float* __restrict__ obuf) {
  const int lane = threadIdx.x & 31, wave = threadIdx.x >> 5;
  const int p = blockIdx.x;                   // b*96 + w
  const int b = p / 96, w = p % 96;
  const int m   = lane & 15;
  const int kb  = (lane & 16) ? 8 : 0;
  const int klo = (lane & 16) ? 8 : 0;
  const unsigned short* vb = vh + (unsigned)p * CC * HH;   // [c][H]

  for (int ci = wave; ci < 16; ci += 8) {     // 2 c-tiles per wave
    v8f acc[6];
    #pragma unroll
    for (int nt = 0; nt < 6; ++nt) acc[nt] = (v8f){};
    #pragma unroll
    for (int k0 = 0; k0 < 96; k0 += 32) {
      FragB A;
      const unsigned short* va = vb + (unsigned)(ci * 16 + m) * HH + k0;
      A.q[0] = *(const uint4*)(va + kb);
      A.q[1] = *(const uint4*)(va + kb + 16);
      #pragma unroll
      for (int nt = 0; nt < 6; ++nt) {
        FragB Bf;
        const unsigned short* ab = ah + ((unsigned)(b * 96 + nt * 16 + m) * 96 + w) * 96 + k0 + klo;
        Bf.q[0] = *(const uint4*)(ab);
        Bf.q[1] = *(const uint4*)(ab + 16);
        acc[nt] = __builtin_amdgcn_wmma_f32_16x16x32_bf16(
                      false, A.v, false, Bf.v, (short)0, acc[nt], false, false);
      }
    }
    #pragma unroll
    for (int nt = 0; nt < 6; ++nt) {
      const unsigned c0  = ci * 16 + ((lane >> 4) << 3);
      const unsigned hq  = nt * 16 + (lane & 15);
      const unsigned idx0 = ((unsigned)b * CC + c0) * (HH * WW) + hq * WW + (unsigned)w;
      #pragma unroll
      for (int v = 0; v < 8; ++v)
        obuf[idx0 + (unsigned)v * (HH * WW)] = acc[nt][v];
    }
  }
}

// ---------------------------------------------------------------------------
// out_w: per (b,h)  O2 = Vw x Aw^T, fused  out = gamma*(O1+O2)+x
// ---------------------------------------------------------------------------
__global__ __launch_bounds__(256) void out_w_kernel(
    const unsigned short* __restrict__ vw, const unsigned short* __restrict__ aw,
    const float* __restrict__ obuf, const float* __restrict__ x,
    const float* __restrict__ gamma, float* __restrict__ outp) {
  const int lane = threadIdx.x & 31, wave = threadIdx.x >> 5;
  const int p = blockIdx.x;                   // b*96 + h
  const int b = p / 96, h = p % 96;
  const int m   = lane & 15;
  const int kb  = (lane & 16) ? 8 : 0;
  const int klo = (lane & 16) ? 8 : 0;
  const unsigned short* vb = vw + (unsigned)p * CC * WW;   // [c][W]
  const float g = gamma[0];

  for (int ci = wave; ci < 16; ci += 8) {
    v8f acc[6];
    #pragma unroll
    for (int nt = 0; nt < 6; ++nt) acc[nt] = (v8f){};
    #pragma unroll
    for (int k0 = 0; k0 < 96; k0 += 32) {
      FragB A;
      const unsigned short* va = vb + (unsigned)(ci * 16 + m) * WW + k0;
      A.q[0] = *(const uint4*)(va + kb);
      A.q[1] = *(const uint4*)(va + kb + 16);
      #pragma unroll
      for (int nt = 0; nt < 6; ++nt) {
        FragB Bf;
        const unsigned short* ab = aw + ((unsigned)(b * 96 + h) * 96 + nt * 16 + m) * 96 + k0 + klo;
        Bf.q[0] = *(const uint4*)(ab);
        Bf.q[1] = *(const uint4*)(ab + 16);
        acc[nt] = __builtin_amdgcn_wmma_f32_16x16x32_bf16(
                      false, A.v, false, Bf.v, (short)0, acc[nt], false, false);
      }
    }
    #pragma unroll
    for (int nt = 0; nt < 6; ++nt) {
      const unsigned c0 = ci * 16 + ((lane >> 4) << 3);
      const unsigned wq = nt * 16 + (lane & 15);
      const unsigned idx0 = ((unsigned)b * CC + c0) * (HH * WW) + (unsigned)h * WW + wq;
      #pragma unroll
      for (int v = 0; v < 8; ++v) {
        const unsigned idx = idx0 + (unsigned)v * (HH * WW);
        outp[idx] = g * (acc[nt][v] + obuf[idx]) + x[idx];
      }
    }
  }
}

// ---------------------------------------------------------------------------
extern "C" void kernel_launch(void* const* d_in, const int* in_sizes, int n_in,
                              void* d_out, int out_size, void* d_ws, size_t ws_size,
                              hipStream_t stream) {
  (void)in_sizes; (void)n_in; (void)out_size; (void)ws_size;
  const float* x    = (const float*)d_in[0];
  const float* wq_h = (const float*)d_in[1];
  const float* bq_h = (const float*)d_in[2];
  const float* wq_w = (const float*)d_in[3];
  const float* bq_w = (const float*)d_in[4];
  const float* wk_h = (const float*)d_in[5];
  const float* bk_h = (const float*)d_in[6];
  const float* wk_w = (const float*)d_in[7];
  const float* bk_w = (const float*)d_in[8];
  const float* wv_h = (const float*)d_in[9];
  const float* bv_h = (const float*)d_in[10];
  const float* wv_w = (const float*)d_in[11];
  const float* bv_w = (const float*)d_in[12];
  const float* gam  = (const float*)d_in[13];

  char* ws = (char*)d_ws;
  size_t off = 0;
  auto carve = [&](size_t bytes) -> char* {
    off = (off + 255) & ~(size_t)255;
    char* pp = ws + off; off += bytes; return pp;
  };
  const size_t NPIX = (size_t)BB * HH * WW;          // 147456
  unsigned short* XT  = (unsigned short*)carve(NPIX * CC * 2);
  unsigned short* WQH = (unsigned short*)carve((size_t)C8 * RLEN * 2);
  unsigned short* WQW = (unsigned short*)carve((size_t)C8 * RLEN * 2);
  unsigned short* WKH = (unsigned short*)carve((size_t)C8 * RLEN * 2);
  unsigned short* WKW = (unsigned short*)carve((size_t)C8 * RLEN * 2);
  unsigned short* WVH = (unsigned short*)carve((size_t)CC * RLEN * 2);
  unsigned short* WVW = (unsigned short*)carve((size_t)CC * RLEN * 2);
  unsigned short* QH  = (unsigned short*)carve(NPIX * C8 * 2);
  unsigned short* QW  = (unsigned short*)carve(NPIX * C8 * 2);
  unsigned short* KH  = (unsigned short*)carve(NPIX * C8 * 2);
  unsigned short* KW  = (unsigned short*)carve(NPIX * C8 * 2);
  unsigned short* VH  = (unsigned short*)carve(NPIX * CC * 2);
  unsigned short* VW  = (unsigned short*)carve(NPIX * CC * 2);
  float*          E   = (float*)carve(NPIX * LCOL * 4);
  unsigned short* AH  = (unsigned short*)carve(NPIX * 96 * 2);
  unsigned short* AW  = (unsigned short*)carve(NPIX * 96 * 2);
  float*          OB  = (float*)carve(NPIX * CC * 4);

  // 1) pack inputs to bf16 / WMMA-friendly layouts
  pack_x_kernel<<<BB * HH * 3 * 8, 256, 0, stream>>>(x, XT);
  pack_w_kernel<<<(C8 * RLEN) / 256, 256, 0, stream>>>(wq_h, WQH, C8);
  pack_w_kernel<<<(C8 * RLEN) / 256, 256, 0, stream>>>(wq_w, WQW, C8);
  pack_w_kernel<<<(C8 * RLEN) / 256, 256, 0, stream>>>(wk_h, WKH, C8);
  pack_w_kernel<<<(C8 * RLEN) / 256, 256, 0, stream>>>(wk_w, WKW, C8);
  pack_w_kernel<<<(CC * RLEN) / 256, 256, 0, stream>>>(wv_h, WVH, CC);
  pack_w_kernel<<<(CC * RLEN) / 256, 256, 0, stream>>>(wv_w, WVW, CC);

  // 2) six convs as implicit WMMA GEMMs (output layouts chosen per consumer)
  const int qkBlocks = BB * HH * (WW / 16) * 1 / 8;        // COUT=32, NB=2 -> 1152
  const int vBlocks  = BB * HH * (WW / 16) * 4 / 8;        // COUT=256, NB=4 -> 4608
  // q_h / k_h : [b][w][h][c]
  conv_wmma_kernel<C8, 1, 2><<<qkBlocks, 256, 0, stream>>>(XT, WQH, bq_h, QH,
      WW * HH * C8, C8, HH * C8, 1u);
  conv_wmma_kernel<C8, 1, 2><<<qkBlocks, 256, 0, stream>>>(XT, WKH, bk_h, KH,
      WW * HH * C8, C8, HH * C8, 1u);
  // q_w / k_w : [b][h][w][c]
  conv_wmma_kernel<C8, 0, 2><<<qkBlocks, 256, 0, stream>>>(XT, WQW, bq_w, QW,
      HH * WW * C8, WW * C8, C8, 1u);
  conv_wmma_kernel<C8, 0, 2><<<qkBlocks, 256, 0, stream>>>(XT, WKW, bk_w, KW,
      HH * WW * C8, WW * C8, C8, 1u);
  // v_h : [b][w][c][H]
  conv_wmma_kernel<CC, 1, 4><<<vBlocks, 256, 0, stream>>>(XT, WVH, bv_h, VH,
      WW * CC * HH, 1u, CC * HH, HH);
  // v_w : [b][h][c][W]
  conv_wmma_kernel<CC, 0, 4><<<vBlocks, 256, 0, stream>>>(XT, WVW, bv_w, VW,
      HH * CC * WW, CC * WW, 1u, WW);

  // 3) logits (async-LDS staged WMMA), diag mask on e_h
  attn_e_kernel<96, 1, 0, 1><<<BB * WW, 256, 0, stream>>>(QH, KH, E);
  attn_e_kernel<1, 96, 96, 0><<<BB * HH, 256, 0, stream>>>(QW, KW, E);

  // 4) joint softmax over 192 logits
  softmax_kernel<<<(unsigned)(NPIX / 8), 256, 0, stream>>>(E, AH, AW);

  // 5) apply attention to V: two WMMA GEMMs, second fuses residual
  out_h_kernel<<<BB * WW, 256, 0, stream>>>(VH, AH, OB);
  out_w_kernel<<<BB * HH, 256, 0, stream>>>(VW, AW, OB, x, gam, (float*)d_out);
}
